// BahdanauAttention_31104153157737
// MI455X (gfx1250) — compile-verified, matrix operand order checked by reading, not run on hardware
//
#include <hip/hip_runtime.h>

// B=2, TQ=TV=1024, D=512, UNITS=128
#define NB   2
#define TQ   1024
#define TV   1024
#define DD   512
#define UU   128

typedef __attribute__((ext_vector_type(16))) __bf16 v16bf;
typedef __attribute__((ext_vector_type(8)))  __bf16 v8bf;
typedef __attribute__((ext_vector_type(4)))  __bf16 v4bf;
typedef __attribute__((ext_vector_type(8)))  float  v8f;

__device__ __forceinline__ float fast_tanh(float x) {
#if __has_builtin(__builtin_amdgcn_tanhf)
  return __builtin_amdgcn_tanhf(x);   // gfx1250 v_tanh_f32
#else
  return tanhf(x);
#endif
}

// Build a 16x32 bf16 WMMA operand fragment from contiguous-K bf16 memory.
// Per ISA 7.12.2: lane half 0 holds K = {0..7, 16..23}, half 1 holds {8..15, 24..31}.
// Each half-run of 8 bf16 is 16 contiguous bytes -> one global_load_b128.
__device__ __forceinline__ v16bf frag_bf16(const __bf16* __restrict__ p, int half) {
  const v8bf lo = *(const v8bf*)(p + (half << 3));        // K = half*8 + 0..7
  const v8bf hi = *(const v8bf*)(p + 16 + (half << 3));   // K = 16 + half*8 + 0..7
  return __builtin_shufflevector(lo, hi, 0, 1, 2, 3, 4, 5, 6, 7,
                                 8, 9, 10, 11, 12, 13, 14, 15);
}

// C/D 16x16 f32: VGPR v, lanes 0-15 -> M=v, lanes 16-31 -> M=8+v, N = lane&15
__device__ __forceinline__ void store_tile(float* __restrict__ C, int row0, int col0,
                                           int ldc, v8f acc, int lane) {
  const int n  = lane & 15;
  const int m0 = (lane >> 4) << 3;
#pragma unroll
  for (int v = 0; v < 8; ++v)
    C[(size_t)(row0 + m0 + v) * ldc + col0 + n] = acc[v];
}

// ---- software-pipelined 16x128 WMMA strip: C[16 x 8*16] += A[16 x K] * B[K x 128] ----
// arowp: lane's A row base (row0 + lane&15), contiguous K, leading dim = LDB (= K here).
// bcolp: lane's B^T col base (col0 + lane&15), contiguous K.
// 2-deep B-fragment rotation + next-step A prefetch: loads for tiles nt+1, nt+2 are in
// flight when WMMA nt issues, so s_wait_loadcnt in front of each WMMA tolerates >0.
template <int KDIM, int LDB>
__device__ __forceinline__ void wmma_strip(const __bf16* __restrict__ arowp,
                                           const __bf16* __restrict__ bcolp,
                                           int half, v8f acc[8]) {
  const size_t NTS = (size_t)16 * LDB;    // col-tile stride in B^T
  v16bf a  = frag_bf16(arowp, half);
  v16bf b0 = frag_bf16(bcolp, half);
  v16bf b1 = frag_bf16(bcolp + NTS, half);
  for (int k0 = 0; k0 < KDIM; k0 += 32) {
    const int kn = (k0 + 32 < KDIM) ? (k0 + 32) : k0;   // clamp on last step
    v16bf a_next = a;
#pragma unroll
    for (int nt = 0; nt < 8; ++nt) {
      v16bf b2;
      if (nt < 6) {
        b2 = frag_bf16(bcolp + (size_t)(nt + 2) * NTS + k0, half);
      } else if (nt == 6) {
        b2 = frag_bf16(bcolp + kn, half);               // next k-step, tile 0
      } else {
        b2 = frag_bf16(bcolp + NTS + kn, half);         // next k-step, tile 1
        a_next = frag_bf16(arowp + kn, half);           // next k-step A
      }
      acc[nt] = __builtin_amdgcn_wmma_f32_16x16x32_bf16(
          false, a, false, b0, (short)0, acc[nt], false, false);
      b0 = b1; b1 = b2;
    }
    a = a_next;
  }
}

// ---------- Pass 0a: f32 -> bf16 rowwise convert (b128 in / b64 out) ----------
__global__ __launch_bounds__(256) void cvt_bf16_kernel(const float* __restrict__ src,
                                                       __bf16* __restrict__ dst) {
  const int i = blockIdx.x * 256 + threadIdx.x;   // one float4 per thread
  const float4 v = ((const float4*)src)[i];
  v4bf o;
  o[0] = (__bf16)v.x; o[1] = (__bf16)v.y; o[2] = (__bf16)v.z; o[3] = (__bf16)v.w;
  *(v4bf*)(dst + (size_t)i * 4) = o;
}

// ---------- Pass 0b: f32 [R x C] -> bf16 [C x R] transpose (32x32 LDS tiles) ----------
__global__ __launch_bounds__(256) void transpose_bf16_kernel(
    const float* __restrict__ src, __bf16* __restrict__ dst,
    int R, int C, size_t src_batch_stride, size_t dst_batch_stride) {
  __shared__ __bf16 tile[32][33];   // +1 pad kills bank conflicts
  const float* __restrict__ s = src + blockIdx.z * src_batch_stride;
  __bf16* __restrict__ d      = dst + blockIdx.z * dst_batch_stride;
  const int tx = threadIdx.x & 31;
  const int ty = threadIdx.x >> 5;  // 0..7
#pragma unroll
  for (int j = 0; j < 4; ++j) {
    const int r = blockIdx.y * 32 + ty + 8 * j;
    tile[ty + 8 * j][tx] = (__bf16)s[(size_t)r * C + blockIdx.x * 32 + tx];
  }
  __syncthreads();
#pragma unroll
  for (int j = 0; j < 4; ++j) {
    const int c = blockIdx.x * 32 + ty + 8 * j;
    d[(size_t)c * R + blockIdx.y * 32 + tx] = tile[tx][ty + 8 * j];
  }
}

// ---------- Pass 1: Q = query@W1, K = value@W2   (M=2048, K=512, N=128) ----------
__global__ __launch_bounds__(256) void bahdanau_proj_kernel(
    const __bf16* __restrict__ qbf, const __bf16* __restrict__ vbf,
    const __bf16* __restrict__ W1t, const __bf16* __restrict__ W2t,
    float* __restrict__ Qp, float* __restrict__ Kp) {
  const __bf16* X; const __bf16* Wt; float* Out;
  if (blockIdx.y == 0) { X = qbf; Wt = W1t; Out = Qp; }
  else                 { X = vbf; Wt = W2t; Out = Kp; }

  const int lane = threadIdx.x & 31;
  const int wave = threadIdx.x >> 5;
  const int half = lane >> 4;
  const int l15  = lane & 15;
  const int row0 = blockIdx.x * 128 + wave * 16;   // 8 waves -> 128 rows/block

  v8f acc[8] = {};
  wmma_strip<DD, DD>(X + (size_t)(row0 + l15) * DD,
                     Wt + (size_t)l15 * DD,        // col0 = 0 (N = 128 = whole strip)
                     half, acc);
#pragma unroll
  for (int nt = 0; nt < 8; ++nt)
    store_tile(Out, row0, nt * 16, UU, acc[nt], lane);
}

// ---------- Pass 2: scores + softmax (one block per (b,q) row) ----------
__global__ __launch_bounds__(256) void bahdanau_score_softmax_kernel(
    const float* __restrict__ Qp, const float* __restrict__ Kp,
    const float* __restrict__ scale, float* __restrict__ attn,
    __bf16* __restrict__ attn_bf) {
  __shared__ float qv[UU];
  __shared__ float sc[UU];
  __shared__ float wred[8];

  const int tid = threadIdx.x;
  const int row = blockIdx.x;            // [0, B*TQ)
  const int b   = row >> 10;

  if (tid < UU) { qv[tid] = Qp[row * UU + tid]; sc[tid] = scale[tid]; }
  __syncthreads();

  const float* __restrict__ Kb = Kp + b * (TV * UU);
  float s[4];
#pragma unroll
  for (int j = 0; j < 4; ++j) {
    const int t = tid + j * 256;
    const float4* __restrict__ kr = (const float4*)(Kb + t * UU);
    float a0 = 0.f, a1 = 0.f, a2 = 0.f, a3 = 0.f;
#pragma unroll 8
    for (int u = 0; u < UU / 4; ++u) {
      const float4 kv = kr[u];
      a0 += sc[4 * u + 0] * fast_tanh(qv[4 * u + 0] + kv.x);
      a1 += sc[4 * u + 1] * fast_tanh(qv[4 * u + 1] + kv.y);
      a2 += sc[4 * u + 2] * fast_tanh(qv[4 * u + 2] + kv.z);
      a3 += sc[4 * u + 3] * fast_tanh(qv[4 * u + 3] + kv.w);
    }
    s[j] = (a0 + a1) + (a2 + a3);
  }

  // row max (wave32 shuffle + cross-wave LDS)
  float m = fmaxf(fmaxf(s[0], s[1]), fmaxf(s[2], s[3]));
#pragma unroll
  for (int off = 16; off; off >>= 1) m = fmaxf(m, __shfl_xor(m, off, 32));
  if ((tid & 31) == 0) wred[tid >> 5] = m;
  __syncthreads();
  float mall = wred[0];
#pragma unroll
  for (int w = 1; w < 8; ++w) mall = fmaxf(mall, wred[w]);
  __syncthreads();

  float e[4]; float lsum = 0.f;
#pragma unroll
  for (int j = 0; j < 4; ++j) { e[j] = __expf(s[j] - mall); lsum += e[j]; }
#pragma unroll
  for (int off = 16; off; off >>= 1) lsum += __shfl_xor(lsum, off, 32);
  if ((tid & 31) == 0) wred[tid >> 5] = lsum;
  __syncthreads();
  float sall = 0.f;
#pragma unroll
  for (int w = 0; w < 8; ++w) sall += wred[w];
  const float inv = 1.0f / sall;

  float*  __restrict__ arow = attn    + (size_t)row * TV;
  __bf16* __restrict__ brow = attn_bf + (size_t)row * TV;
#pragma unroll
  for (int j = 0; j < 4; ++j) {
    const float w = e[j] * inv;
    arow[tid + j * 256] = w;
    brow[tid + j * 256] = (__bf16)w;     // free bf16 copy for the context GEMM
  }
}

// ---------- Pass 3: context = attn @ value   (per batch: 1024x1024 @ 1024x512) ----------
__global__ __launch_bounds__(256) void bahdanau_context_kernel(
    const __bf16* __restrict__ attn_bf, const __bf16* __restrict__ Vt,
    float* __restrict__ ctx) {
  const int b = blockIdx.z;
  const __bf16* __restrict__ A  = attn_bf + (size_t)b * TQ * TV;
  const __bf16* __restrict__ Bt = Vt      + (size_t)b * DD * TV;
  float* __restrict__ C         = ctx     + (size_t)b * TQ * DD;

  const int lane = threadIdx.x & 31;
  const int wave = threadIdx.x >> 5;
  const int half = lane >> 4;
  const int l15  = lane & 15;
  const int row0 = blockIdx.x * 128 + wave * 16;
  const int col0 = blockIdx.y * 128;

  v8f acc[8] = {};
  wmma_strip<TV, TV>(A  + (size_t)(row0 + l15) * TV,
                     Bt + (size_t)(col0 + l15) * TV,
                     half, acc);
#pragma unroll
  for (int nt = 0; nt < 8; ++nt)
    store_tile(C, row0, col0 + nt * 16, DD, acc[nt], lane);
}

extern "C" void kernel_launch(void* const* d_in, const int* in_sizes, int n_in,
                              void* d_out, int out_size, void* d_ws, size_t ws_size,
                              hipStream_t stream) {
  const float* query = (const float*)d_in[0];   // [B,TQ,D]
  const float* value = (const float*)d_in[1];   // [B,TV,D]
  const float* W1    = (const float*)d_in[2];   // [D,U]
  const float* W2    = (const float*)d_in[3];   // [D,U]
  const float* scale = (const float*)d_in[4];   // [U]

  float* out  = (float*)d_out;
  float* ctx  = out;                              // [B,TQ,D]   (first output)
  float* attn = out + (size_t)NB * TQ * DD;       // [B,TQ,TV]  (second output)

  // ---- workspace layout ----
  char* ws = (char*)d_ws;
  float*  Qp  = (float*)ws;                        ws += (size_t)NB * TQ * UU * 4;  // 1 MB
  float*  Kp  = (float*)ws;                        ws += (size_t)NB * TV * UU * 4;  // 1 MB
  __bf16* qbf = (__bf16*)ws;                       ws += (size_t)NB * TQ * DD * 2;  // 2 MB
  __bf16* vbf = (__bf16*)ws;                       ws += (size_t)NB * TV * DD * 2;  // 2 MB
  __bf16* Vt  = (__bf16*)ws;                       ws += (size_t)NB * DD * TV * 2;  // 2 MB
  __bf16* W1t = (__bf16*)ws;                       ws += (size_t)UU * DD * 2;       // 128 KB
  __bf16* W2t = (__bf16*)ws;                       ws += (size_t)UU * DD * 2;       // 128 KB
  __bf16* abf = (__bf16*)ws;                       /* NB*TQ*TV*2 = 4 MB */

  // Pass 0: precision staging (memory-bound, ~10 MB total => ~1us at 23.3 TB/s)
  cvt_bf16_kernel<<<(NB * TQ * DD) / (4 * 256), 256, 0, stream>>>(query, qbf);
  cvt_bf16_kernel<<<(NB * TV * DD) / (4 * 256), 256, 0, stream>>>(value, vbf);
  transpose_bf16_kernel<<<dim3(DD / 32, TV / 32, NB), 256, 0, stream>>>(
      value, Vt, TV, DD, (size_t)TV * DD, (size_t)DD * TV);
  transpose_bf16_kernel<<<dim3(UU / 32, DD / 32, 1), 256, 0, stream>>>(
      W1, W1t, DD, UU, 0, 0);
  transpose_bf16_kernel<<<dim3(UU / 32, DD / 32, 1), 256, 0, stream>>>(
      W2, W2t, DD, UU, 0, 0);

  // Pass 1: both projections (grid.y selects Q vs K path)
  bahdanau_proj_kernel<<<dim3((NB * TQ) / 128, 2), 256, 0, stream>>>(
      qbf, vbf, W1t, W2t, Qp, Kp);

  // Pass 2: additive scores + softmax (dominant phase: 268M v_tanh_f32 + FMA)
  bahdanau_score_softmax_kernel<<<NB * TQ, 256, 0, stream>>>(Qp, Kp, scale, attn, abf);

  // Pass 3: context GEMM
  bahdanau_context_kernel<<<dim3(TQ / 128, DD / 128, NB), 256, 0, stream>>>(
      abf, Vt, ctx);
}
